// RegiNet_smallViT_6262062318011
// MI455X (gfx1250) — matrix operand order repaired and checked by reading, not compiled, not run on hardware
//
#include <hip/hip_runtime.h>
#include <cmath>

typedef __attribute__((ext_vector_type(16))) _Float16 v16h;
typedef __attribute__((ext_vector_type(8)))  _Float16 v8h;
typedef __attribute__((ext_vector_type(8)))  float    v8f;

#define VOL   2097152L          // 128^3
#define HW    16384             // 128*128
#define NP    96                // padded token count (65 -> 96 = 3*32)

// ---------------------------------------------------------------- utilities
__global__ void zero_kernel(float* __restrict__ p, int n) {
    int i = blockIdx.x * blockDim.x + threadIdx.x;
    if (i < n) p[i] = 0.f;
}

__device__ __forceinline__ float gelu_f(float x) {
    const float c = 0.7978845608028654f;     // sqrt(2/pi)
    float u = c * (x + 0.044715f * x * x * x);
    return 0.5f * x * (1.f + tanhf(u));
}

// ---------------------------------------------------------------- conv3d (direct, SAME)
__global__ void conv3d_kernel(const float* __restrict__ in, const float* __restrict__ w,
                              const float* __restrict__ bias, float* __restrict__ out,
                              int CI, int CO) {
    long tid = (long)blockIdx.x * 256 + threadIdx.x;
    if (tid >= (long)CO * VOL) return;
    int  co  = (int)(tid >> 21);
    long rem = tid & (VOL - 1);
    int z = (int)(rem >> 14);
    int y = (int)((rem >> 7) & 127);
    int x = (int)(rem & 127);
    float acc = bias[co];
    for (int ci = 0; ci < CI; ++ci) {
        const float* inp = in + (long)ci * VOL;
        const float* wp  = w + (long)(co * CI + ci) * 27;
        for (int dz = -1; dz <= 1; ++dz) {
            int zz = z + dz; if (zz < 0 || zz > 127) continue;
            for (int dy = -1; dy <= 1; ++dy) {
                int yy = y + dy; if (yy < 0 || yy > 127) continue;
                const float* row = inp + ((long)zz << 14) + (yy << 7);
                const float* wr  = wp + (dz + 1) * 9 + (dy + 1) * 3;
                for (int dx = -1; dx <= 1; ++dx) {
                    int xx = x + dx; if (xx < 0 || xx > 127) continue;
                    acc += wr[dx + 1] * row[xx];
                }
            }
        }
    }
    out[tid] = acc;
}

// ---------------------------------------------------------------- batch-norm (stats + apply + relu)
__global__ void bn_stats_kernel(const float* __restrict__ x, float* __restrict__ stats) {
    __shared__ float s1[256], s2[256];
    long tid = (long)blockIdx.x * 256 + threadIdx.x;     // grid sized exactly C*VOL/256
    float v = x[tid];
    s1[threadIdx.x] = v; s2[threadIdx.x] = v * v;
    __syncthreads();
    for (int st = 128; st > 0; st >>= 1) {
        if (threadIdx.x < st) { s1[threadIdx.x] += s1[threadIdx.x + st];
                                s2[threadIdx.x] += s2[threadIdx.x + st]; }
        __syncthreads();
    }
    if (threadIdx.x == 0) {
        int c = (int)(tid >> 21);
        atomicAdd(&stats[2 * c],     s1[0]);
        atomicAdd(&stats[2 * c + 1], s2[0]);
    }
}

__global__ void bn_apply_kernel(float* __restrict__ x, const float* __restrict__ stats,
                                const float* __restrict__ g, const float* __restrict__ beta,
                                int C) {
    long tid = (long)blockIdx.x * 256 + threadIdx.x;
    if (tid >= (long)C * VOL) return;
    int c = (int)(tid >> 21);
    const float invV = 1.0f / 2097152.0f;
    float mean = stats[2 * c] * invV;
    float var  = stats[2 * c + 1] * invV - mean * mean;
    float v = (x[tid] - mean) * rsqrtf(var + 1e-5f) * g[c] + beta[c];
    x[tid] = v > 0.f ? v : 0.f;
}

// ---------------------------------------------------------------- residual: x3d[c] = x + conv[c]
__global__ void residual3_kernel(const float* __restrict__ x, const float* __restrict__ conv,
                                 float* __restrict__ x3d) {
    long tid = (long)blockIdx.x * 256 + threadIdx.x;
    if (tid >= 3 * VOL) return;
    x3d[tid] = x[tid & (VOL - 1)] + conv[tid];
}

// ---------------------------------------------------------------- rodrigues pose
__global__ void pose_kernel(const float* __restrict__ th, float* __restrict__ T) {
    if (threadIdx.x != 0 || blockIdx.x != 0) return;
    float r0 = th[0], r1 = th[1], r2 = th[2];
    float ang = sqrtf(r0 * r0 + r1 * r1 + r2 * r2 + 1e-12f);
    float kx = r0 / ang, ky = r1 / ang, kz = r2 / ang;
    float c = cosf(ang), s = sinf(ang), ic = 1.f - c;
    T[0]  = c + ic * kx * kx;      T[1]  = -s * kz + ic * kx * ky; T[2]  = s * ky + ic * kx * kz;  T[3]  = th[3];
    T[4]  = s * kz + ic * ky * kx; T[5]  = c + ic * ky * ky;       T[6]  = -s * kx + ic * ky * kz; T[7]  = th[4];
    T[8]  = -s * ky + ic * kz * kx;T[9]  = s * kx + ic * kz * ky;  T[10] = c + ic * kz * kz;       T[11] = th[5];
}

// ---------------------------------------------------------------- transform + trilinear + K-sum
__global__ void sample_kernel(const float* __restrict__ x3d, const float* __restrict__ grid,
                              const float* __restrict__ T, float* __restrict__ x2d) {
    int p = blockIdx.x * blockDim.x + threadIdx.x;
    if (p >= HW) return;
    float a0 = 0.f, a1 = 0.f, a2 = 0.f;
    for (int k = 0; k < 64; ++k) {
        const float* g = grid + ((long)p * 64 + k) * 4;
        float g0 = g[0], g1 = g[1], g2 = g[2], g3 = g[3];
        float tx = g0 * T[0] + g1 * T[1] + g2 * T[2]  + g3 * T[3];
        float ty = g0 * T[4] + g1 * T[5] + g2 * T[6]  + g3 * T[7];
        float tz = g0 * T[8] + g1 * T[9] + g2 * T[10] + g3 * T[11];
        float gx = 128.f * (tx * 0.5f + 0.5f);
        float gy = 128.f * (ty * 0.5f + 0.5f);
        float gz = 128.f * (tz * 0.5f + 0.5f);
        if (!(gx >= 0.f && gx <= 128.f && gy >= 0.f && gy <= 128.f && gz >= 0.f && gz <= 128.f))
            continue;
        float fx = floorf(gx), fy = floorf(gy), fz = floorf(gz);
        float x0 = fminf(fmaxf(fx, 0.f), 127.f),       x1 = fminf(fmaxf(fx + 1.f, 0.f), 127.f);
        float y0 = fminf(fmaxf(fy, 0.f), 127.f),       y1 = fminf(fmaxf(fy + 1.f, 0.f), 127.f);
        float z0 = fminf(fmaxf(fz, 0.f), 127.f),       z1 = fminf(fmaxf(fz + 1.f, 0.f), 127.f);
        float wz[2] = { z1 - gz, gz - z0 };
        float wy[2] = { y1 - gy, gy - y0 };
        float wx[2] = { x1 - gx, gx - x0 };
        int iz[2] = { (int)z0, (int)z1 };
        int iy[2] = { (int)y0, (int)y1 };
        int ix[2] = { (int)x0, (int)x1 };
        for (int a = 0; a < 2; ++a)
            for (int b = 0; b < 2; ++b)
                for (int c = 0; c < 2; ++c) {
                    float w = wz[a] * wy[b] * wx[c];
                    long idx = ((long)iz[a] << 14) + (iy[b] << 7) + ix[c];
                    a0 += w * x3d[idx];
                    a1 += w * x3d[VOL + idx];
                    a2 += w * x3d[2 * VOL + idx];
                }
    }
    x2d[p] = a0; x2d[HW + p] = a1; x2d[2 * HW + p] = a2;
}

// ---------------------------------------------------------------- patch extraction (64 x 768)
__global__ void im2patch_kernel(const float* __restrict__ img, float* __restrict__ patches,
                                int bcast) {
    int tid = blockIdx.x * blockDim.x + threadIdx.x;
    if (tid >= 64 * 768) return;
    int patch = tid / 768, f = tid % 768;
    int c = f % 3, pcidx = f / 3;
    int pr = pcidx / 16, pc = pcidx % 16;
    int ph = patch / 8,  pw = patch % 8;
    int row = ph * 16 + pr, col = pw * 16 + pc;
    patches[tid] = img[(bcast ? 0 : c) * HW + row * 128 + col];
}

// ---------------------------------------------------------------- token assembly (NP x 128, zero pads)
__global__ void tokens_kernel(const float* __restrict__ pe, const float* __restrict__ cls,
                              const float* __restrict__ pos, float* __restrict__ t) {
    int tid = blockIdx.x * blockDim.x + threadIdx.x;
    if (tid >= NP * 128) return;
    int i = tid / 128, j = tid % 128;
    if (i >= 65) { t[tid] = 0.f; return; }
    t[tid] = ((i == 0) ? cls[j] : pe[(i - 1) * 128 + j]) + pos[tid];
}

// ---------------------------------------------------------------- layernorm over dim 128 (NP rows)
__global__ void ln_kernel(const float* __restrict__ t, const float* __restrict__ g,
                          const float* __restrict__ b, float* __restrict__ h) {
    int row = blockIdx.x, j = threadIdx.x;
    __shared__ float s[128];
    float v = t[row * 128 + j];
    s[j] = v; __syncthreads();
    for (int st = 64; st > 0; st >>= 1) { if (j < st) s[j] += s[j + st]; __syncthreads(); }
    float mean = s[0] * (1.f / 128.f);
    __syncthreads();
    float d = v - mean;
    s[j] = d * d; __syncthreads();
    for (int st = 64; st > 0; st >>= 1) { if (j < st) s[j] += s[j + st]; __syncthreads(); }
    float var = s[0] * (1.f / 128.f);
    h[row * 128 + j] = d * rsqrtf(var + 1e-5f) * g[j] + b[j];
}

// ---------------------------------------------------------------- softmax over first 65 cols of NP; zero pads
__global__ void softmax_kernel(float* __restrict__ att) {
    int row = blockIdx.x, j = threadIdx.x;     // 128 threads
    __shared__ float s[128];
    float v = (j < 65) ? att[row * NP + j] : -3.4e38f;
    s[j] = v; __syncthreads();
    for (int st = 64; st > 0; st >>= 1) { if (j < st) s[j] = fmaxf(s[j], s[j + st]); __syncthreads(); }
    float m = s[0]; __syncthreads();
    float e = (j < 65) ? expf(v - m) : 0.f;
    s[j] = e; __syncthreads();
    for (int st = 64; st > 0; st >>= 1) { if (j < st) s[j] += s[j + st]; __syncthreads(); }
    if (j < 65) att[row * NP + j] = e / s[0];
    else if (j < NP) att[row * NP + j] = 0.f;        // zero pad keys for the K=NP att@V GEMM
}

// ---------------------------------------------------------------- classifier head (cls row only)
__global__ void head_kernel(const float* __restrict__ h0, const float* __restrict__ w,
                            const float* __restrict__ b, float* __restrict__ out) {
    int j = blockIdx.x * blockDim.x + threadIdx.x;
    if (j >= 1000) return;
    float acc = b[j];
    for (int i = 0; i < 128; ++i) acc += h0[i] * w[i * 1000 + j];
    out[j] = acc;
}

// ---------------------------------------------------------------- WMMA GEMM, double-buffered LDS
// Static requirements: M % 16 == 0 (rows exist in padded buffers), K % 32 == 0,
// ALN => N % 64 == 0 (no column guards anywhere).
// Block = 128 threads (4 waves). Tile: 16 (M) x 64 (N); wave w owns cols w*16..w*16+15.
struct StageRegs { float4 a; float4 b[4]; };

template<bool TRANSB, bool ALN>
__device__ __forceinline__ StageRegs stage_load(const float* __restrict__ Az, int lda,
                                                const float* __restrict__ Bz, int ldb,
                                                int rowBase, int colBase0, int N, int k0, int tid) {
    StageRegs s;
    int t4 = tid * 4;                         // 0..508, covers the 16x32 A tile exactly
    int r = t4 >> 5, kk = t4 & 31;
    const float* ap = Az + (long)(rowBase + r) * lda + k0 + kk;
    __builtin_prefetch(ap + 32, 0, 0);        // next K tile
    s.a = *(const float4*)ap;
#pragma unroll
    for (int j = 0; j < 4; ++j) {
        int linear = (tid + j * 128) * 4;     // 0..2044, row-major [32][64] order
        int kb = linear >> 6, nn = linear & 63;
        int n  = colBase0 + nn;
        if (!TRANSB && (ALN || n + 3 < N)) {
            s.b[j] = *(const float4*)(Bz + (long)(k0 + kb) * ldb + n);
        } else {
            s.b[j].x = (ALN || n     < N) ? (TRANSB ? Bz[(long)(n    ) * ldb + k0 + kb] : Bz[(long)(k0 + kb) * ldb + n    ]) : 0.f;
            s.b[j].y = (ALN || n + 1 < N) ? (TRANSB ? Bz[(long)(n + 1) * ldb + k0 + kb] : Bz[(long)(k0 + kb) * ldb + n + 1]) : 0.f;
            s.b[j].z = (ALN || n + 2 < N) ? (TRANSB ? Bz[(long)(n + 2) * ldb + k0 + kb] : Bz[(long)(k0 + kb) * ldb + n + 2]) : 0.f;
            s.b[j].w = (ALN || n + 3 < N) ? (TRANSB ? Bz[(long)(n + 3) * ldb + k0 + kb] : Bz[(long)(k0 + kb) * ldb + n + 3]) : 0.f;
        }
    }
    return s;
}

__device__ __forceinline__ void stage_store(const StageRegs& s, _Float16* __restrict__ As,
                                            _Float16* __restrict__ Bs, int tid) {
    int t4 = tid * 4;
    int r = t4 >> 5, kk = t4 & 31;
    _Float16* d = As + r * 32 + kk;
    d[0] = (_Float16)s.a.x; d[1] = (_Float16)s.a.y;
    d[2] = (_Float16)s.a.z; d[3] = (_Float16)s.a.w;
#pragma unroll
    for (int j = 0; j < 4; ++j) {
        int linear = (tid + j * 128) * 4;
        int kb = linear >> 6, nn = linear & 63;
        Bs[(nn    ) * 32 + kb] = (_Float16)s.b[j].x;   // transposed: Bs[n][k]
        Bs[(nn + 1) * 32 + kb] = (_Float16)s.b[j].y;
        Bs[(nn + 2) * 32 + kb] = (_Float16)s.b[j].z;
        Bs[(nn + 3) * 32 + kb] = (_Float16)s.b[j].w;
    }
}

template<bool TRANSB, bool ALN, int ACT, bool HB, bool HC>
__global__ void __launch_bounds__(128)
gemm_wmma_kernel(const float* __restrict__ A, int lda, long sAz,
                 const float* __restrict__ B, int ldb, long sBz,
                 float* __restrict__ D, int ldd, long sDz,
                 const float* __restrict__ bias, const float* __restrict__ Cres,
                 int N, int K, float alpha) {
    __shared__ __align__(16) _Float16 As[2][16 * 32];
    __shared__ __align__(16) _Float16 Bs[2][64 * 32];
    int tid  = threadIdx.x;
    int wave = tid >> 5, lane = tid & 31;
    int hi = lane >> 4, lo = lane & 15;
    int rowBase  = blockIdx.x * 16;
    int colBase0 = blockIdx.y * 64;
    const float* Az = A + (long)blockIdx.z * sAz;
    const float* Bz = B + (long)blockIdx.z * sBz;
    float*       Dz = D + (long)blockIdx.z * sDz;

    // prologue: stage first tile into buffer 0
    StageRegs sr = stage_load<TRANSB, ALN>(Az, lda, Bz, ldb, rowBase, colBase0, N, 0, tid);
    stage_store(sr, As[0], Bs[0], tid);

    v8f acc = {};
    int p = 0;
    for (int k0 = 0; k0 < K; k0 += 32) {
        __syncthreads();                      // tile p visible; buffer p^1 free
        bool hasNext = (k0 + 32 < K);
        if (hasNext)                          // issue next tile's global loads early
            sr = stage_load<TRANSB, ALN>(Az, lda, Bz, ldb, rowBase, colBase0, N, k0 + 32, tid);
        // fragments from LDS (16B ds_load_b128 reads matching ISA lane layouts)
        // A 16x32: lane(hi,lo) needs k = {hi*8..+7} and {16+hi*8..+7} of row lo
        const v8h* apv = (const v8h*)(As[p] + lo * 32 + hi * 8);
        v8h a0 = apv[0], a1 = apv[2];
        // B 32x16: lane(hi,lo) needs k = {hi*8..+7} and {16+hi*8..+7} of col (wave*16+lo)
        const v8h* bpv = (const v8h*)(Bs[p] + (wave * 16 + lo) * 32 + hi * 8);
        v8h b0 = bpv[0], b1 = bpv[2];
        v16h af, bf;
#pragma unroll
        for (int e = 0; e < 8; ++e) {
            af[e] = a0[e]; af[e + 8] = a1[e];
            bf[e] = b0[e]; bf[e + 8] = b1[e];
        }
        acc = __builtin_amdgcn_wmma_f32_16x16x32_f16(false, af, false, bf,
                                                     (short)0, acc, false, false);
        if (hasNext)                          // convert + ds_store after the WMMA
            stage_store(sr, As[p ^ 1], Bs[p ^ 1], tid);
        p ^= 1;
    }
    // ---- epilogue (col guard only in unaligned instantiations; rows are padded)
    int n = colBase0 + wave * 16 + lo;
#pragma unroll
    for (int r = 0; r < 8; ++r) {
        int row = rowBase + r + 8 * hi;       // C/D layout: VGPR r, hi lanes hold M=r+8
        if (ALN || n < N) {
            float v = acc[r] * alpha;
            if (HB) v += bias[n];
            if (HC) v += Cres[(long)row * ldd + n];
            if (ACT == 1) v = gelu_f(v);
            Dz[(long)row * ldd + n] = v;
        }
    }
}

// ================================================================ host orchestration
extern "C" void kernel_launch(void* const* d_in, const int* in_sizes, int n_in,
                              void* d_out, int out_size, void* d_ws, size_t ws_size,
                              hipStream_t stream) {
    (void)in_sizes; (void)n_in; (void)out_size; (void)ws_size;
    auto IN = [&](int i) { return (const float*)d_in[i]; };

    // ---- workspace carve-out (floats)
    float* W = (float*)d_ws;
    size_t off = 0;
    auto alloc = [&](size_t n) -> float* { float* p = W + off; off += (n + 31) & ~size_t(31); return p; };
    float* stats   = alloc(64);
    float* T34     = alloc(16);
    float* x2d     = alloc(3 * HW);
    float* patches = alloc(64 * 768);
    float* pe      = alloc(64 * 128);
    float* tok     = alloc(NP * 128);
    float* hbuf    = alloc(NP * 128);
    float* qkv     = alloc(NP * 384);
    float* att     = alloc(4 * NP * NP);
    float* obuf    = alloc(NP * 128);
    float* m1      = alloc(NP * 256);
    size_t smallEnd = off;
    float* bufA    = alloc(16 * VOL);
    float* bufB    = alloc(16 * VOL);
    float* x3d     = alloc(3 * VOL);

    // deterministic pads: zero the small scratch region every call
    zero_kernel<<<dim3((unsigned)((smallEnd + 255) / 256)), 256, 0, stream>>>(W, (int)smallEnd);

    // ---- input index map (setup_inputs order, depth-first)
    // 0:x 1:y 2:theta 3:grid | 4-8:conv w | 9-13:conv b | 14-17:g | 18-21:beta
    // 22..62: vit_x params | 63..103: vit_y params
    auto conv = [&](const float* in, float* out, int idx, int CI, int CO) {
        long total = (long)CO * VOL;
        conv3d_kernel<<<dim3((unsigned)((total + 255) / 256)), 256, 0, stream>>>(
            in, IN(4 + idx), IN(9 + idx), out, CI, CO);
    };
    auto bn = [&](float* buf, int idx, int C) {
        zero_kernel<<<1, 64, 0, stream>>>(stats, 2 * C);
        bn_stats_kernel<<<dim3((unsigned)((long)C * VOL / 256)), 256, 0, stream>>>(buf, stats);
        bn_apply_kernel<<<dim3((unsigned)(((long)C * VOL + 255) / 256)), 256, 0, stream>>>(
            buf, stats, IN(14 + idx), IN(18 + idx), C);
    };

    // ---- conv3d stack + residual
    const float* x = IN(0);
    conv(x,    bufA, 0, 1, 4);   bn(bufA, 0, 4);
    conv(bufA, bufB, 1, 4, 8);   bn(bufB, 1, 8);
    conv(bufB, bufA, 2, 8, 16);  bn(bufA, 2, 16);
    conv(bufA, bufB, 3, 16, 8);  bn(bufB, 3, 8);
    conv(bufB, bufA, 4, 8, 3);
    residual3_kernel<<<dim3((unsigned)((3 * VOL + 255) / 256)), 256, 0, stream>>>(x, bufA, x3d);

    // ---- pose + trilinear projection
    pose_kernel<<<1, 32, 0, stream>>>(IN(2), T34);
    sample_kernel<<<dim3(HW / 256), 256, 0, stream>>>(x3d, IN(3), T34, x2d);

    // ---- specialized WMMA GEMM launches (M%16==0, K%32==0 by construction)
#define GEMM(TB, ALNB, ACTV, HBB, HCC, Ap, lda_, sAz_, Bp, ldb_, sBz_, Dp, ldd_, sDz_, biasp, cresp, M_, N_, K_, alpha_, Z_) \
    do { dim3 g_((unsigned)((M_) / 16), (unsigned)(((N_) + 63) / 64), (unsigned)(Z_));                                        \
         gemm_wmma_kernel<TB, ALNB, ACTV, HBB, HCC><<<g_, 128, 0, stream>>>(                                                  \
             Ap, lda_, sAz_, Bp, ldb_, sBz_, Dp, ldd_, sDz_, biasp, cresp, N_, K_, alpha_); } while (0)

    // ---- one ViT forward
    auto run_vit = [&](const float* img, int bcast, int base, float* outp) {
        auto P = [&](int i) { return (const float*)d_in[base + i]; };
        auto L = [&](int l, int j) { return (const float*)d_in[base + 4 + l * 11 + j]; };
        im2patch_kernel<<<dim3((64 * 768 + 255) / 256), 256, 0, stream>>>(img, patches, bcast);
        GEMM(false, true, 0, true, false, patches, 768, 0L, P(0), 128, 0L,
             pe, 128, 0L, P(1), nullptr, 64, 128, 768, 1.f, 1);
        tokens_kernel<<<dim3((NP * 128 + 255) / 256), 256, 0, stream>>>(pe, P(3), P(2), tok);
        for (int l = 0; l < 3; ++l) {
            ln_kernel<<<NP, 128, 0, stream>>>(tok, L(l, 0), L(l, 1), hbuf);
            GEMM(false, true, 0, false, false, hbuf, 128, 0L, L(l, 2), 384, 0L,
                 qkv, 384, 0L, nullptr, nullptr, NP, 384, 128, 1.f, 1);
            // per-head Q @ K^T (B transposed), scaled 1/sqrt(32); head h -> att + h*NP*NP
            GEMM(true, false, 0, false, false, qkv, 384, 32L, qkv + 128, 384, 32L,
                 att, NP, (long)NP * NP, nullptr, nullptr, NP, NP, 32, 0.17677669529663687f, 4);
            softmax_kernel<<<4 * NP, 128, 0, stream>>>(att);
            // att @ V -> obuf columns h*32..h*32+31  (K=NP, pad keys zeroed by softmax)
            GEMM(false, false, 0, false, false, att, NP, (long)NP * NP, qkv + 256, 384, 32L,
                 obuf, 128, 32L, nullptr, nullptr, NP, 32, NP, 1.f, 4);
            // t = t + o @ out_w + out_b
            GEMM(false, true, 0, true, true, obuf, 128, 0L, L(l, 3), 128, 0L,
                 tok, 128, 0L, L(l, 4), tok, NP, 128, 128, 1.f, 1);
            ln_kernel<<<NP, 128, 0, stream>>>(tok, L(l, 5), L(l, 6), hbuf);
            GEMM(false, true, 1, true, false, hbuf, 128, 0L, L(l, 7), 256, 0L,
                 m1, 256, 0L, L(l, 8), nullptr, NP, 256, 128, 1.f, 1);
            GEMM(false, true, 0, true, true, m1, 256, 0L, L(l, 9), 128, 0L,
                 tok, 128, 0L, L(l, 10), tok, NP, 128, 256, 1.f, 1);
        }
        ln_kernel<<<NP, 128, 0, stream>>>(tok, P(37), P(38), hbuf);
        head_kernel<<<4, 256, 0, stream>>>(hbuf, P(39), P(40), outp);
    };

    float* out = (float*)d_out;
    run_vit(x2d,   0, 22, out);          // x_out
    run_vit(IN(1), 1, 63, out + 1000);   // y_out
#undef GEMM
}
